// TransformerLayer_88158498718040
// MI455X (gfx1250) — matrix operand order repaired and checked by reading, not compiled, not run on hardware
//
#include <hip/hip_runtime.h>

// ---------------------------------------------------------------------------
// Transformer layer for MI455X (gfx1250): bf16 WMMA GEMMs + flash attention.
// B=4, S=2048, D=1024, fp32 in/out, bf16 matrix math with f32 accumulation.
// 32-bit addressing, 128-bit fragment loads, double-buffered GEMM staging,
// global_prefetch hints for the next tiles.
// ---------------------------------------------------------------------------

typedef __attribute__((ext_vector_type(16))) __bf16        v16bf;
typedef __attribute__((ext_vector_type(8)))  float         v8f;
typedef __attribute__((ext_vector_type(4)))  unsigned int  v4u;
typedef __attribute__((ext_vector_type(2)))  unsigned int  v2u;

#define DEV_INLINE __device__ __forceinline__

struct F8 { v4u lo, hi; };                      // 8 VGPRs = 16 bf16

DEV_INLINE unsigned short f32_to_bf16(float f) {
    unsigned int u = __float_as_uint(f);
    unsigned int r = u + 0x7FFFu + ((u >> 16) & 1u);   // round-to-nearest-even
    return (unsigned short)(r >> 16);
}
DEV_INLINE unsigned int pack_bf16x2(float a, float b) {
    return (unsigned int)f32_to_bf16(a) | ((unsigned int)f32_to_bf16(b) << 16);
}

// A-matrix (16x32 bf16), ISA 7.12.2: lane-half kh; VGPR0..3 hold K = kh*8+{0..7},
// VGPR4..7 hold K = 16+kh*8+{0..7}  -> two contiguous 16B runs.
DEV_INLINE F8 load_afrag(const unsigned short* p, int kh) {
    F8 f;
    f.lo = *reinterpret_cast<const v4u*>(p + kh * 8);
    f.hi = *reinterpret_cast<const v4u*>(p + 16 + kh * 8);
    return f;
}
// B-matrix (32x16 bf16): lane-half kh holds K = kh*8+{0..7}, 16+kh*8+{0..7} (matches A).
DEV_INLINE F8 load_bfrag(const unsigned short* p, int kh) {
    F8 f;
    f.lo = *reinterpret_cast<const v4u*>(p + kh * 8);
    f.hi = *reinterpret_cast<const v4u*>(p + 16 + kh * 8);
    return f;
}

DEV_INLINE v8f wmma_bf16(const F8& a, const F8& b, v8f c) {
    return __builtin_amdgcn_wmma_f32_16x16x32_bf16(
        false, __builtin_bit_cast(v16bf, a),
        false, __builtin_bit_cast(v16bf, b),
        (short)0, c, false, false);
}

constexpr int GEMM_LDSTR = 40;                  // bf16 stride (mult of 8)

// Stage ITERS*256 float4s of a [rows x 32] fp32 tile into LDS as bf16.
template<int ITERS>
DEV_INLINE void stage_tile(const float* __restrict__ src, unsigned int K,
                           unsigned int kc, unsigned short* __restrict__ dst,
                           unsigned int tid) {
#pragma unroll
    for (int i = 0; i < ITERS; ++i) {
        unsigned int idx = i * 256u + tid;      // float4 index
        unsigned int r = idx >> 3, c4 = idx & 7;
        float4 v = *reinterpret_cast<const float4*>(src + r * K + kc + c4 * 4);
        v2u p = { pack_bf16x2(v.x, v.y), pack_bf16x2(v.z, v.w) };
        *reinterpret_cast<v2u*>(&dst[r * GEMM_LDSTR + c4 * 4]) = p;
    }
}

// ---------------------------------------------------------------------------
// GEMM: C[M,N] = act( scale * A[M,K] @ W[N,K]^T + bias )
// OUTMODE: 0 = f32 row-major, 1 = bf16 row-major, 2 = bf16 transposed [N,M]
// 256 threads (8 waves), 128x64 tile, wave = 32x32 sub-tile, double-buffered.
// ---------------------------------------------------------------------------
template<int OUTMODE, bool RELU, bool HASBIAS>
__global__ void __launch_bounds__(256)
gemm_bf16_kernel(const float* __restrict__ A, const float* __restrict__ W,
                 const float* __restrict__ bias, void* __restrict__ outp,
                 unsigned int M, unsigned int N, unsigned int K, float scale) {
    __shared__ __align__(16) unsigned short sA[2][128 * GEMM_LDSTR];
    __shared__ __align__(16) unsigned short sW[2][64 * GEMM_LDSTR];

    const unsigned int tid  = threadIdx.x;
    const int wave = tid >> 5;
    const int lane = tid & 31;
    const int kh   = lane >> 4;
    const int ml   = lane & 15;

    const unsigned int m0 = blockIdx.y * 128u;
    const unsigned int n0 = blockIdx.x * 64u;
    const int wr = wave >> 1;
    const int wc = wave & 1;

    const float* Abase = A + m0 * K;
    const float* Wbase = W + n0 * K;

    v8f acc[2][2];
#pragma unroll
    for (int mi = 0; mi < 2; ++mi)
#pragma unroll
        for (int ni = 0; ni < 2; ++ni) acc[mi][ni] = (v8f){};

    // Prologue: stage tile 0 into buffer 0
    stage_tile<4>(Abase, K, 0, sA[0], tid);
    stage_tile<2>(Wbase, K, 0, sW[0], tid);
    __syncthreads();

    const unsigned int ntiles = K >> 5;
    for (unsigned int t = 0; t < ntiles; ++t) {
        const int cur = (int)(t & 1u);
        const int nxt = cur ^ 1;

        // Stage next tile into the other buffer (overlaps with WMMA below)
        if (t + 1 < ntiles) {
            unsigned int kc1 = (t + 1) << 5;
            stage_tile<4>(Abase, K, kc1, sA[nxt], tid);
            stage_tile<2>(Wbase, K, kc1, sW[nxt], tid);
        }
        // Prefetch tile t+2 into cache (global_prefetch_b8)
        if (t + 2 < ntiles) {
            unsigned int kp = (t + 2) << 5;
            __builtin_prefetch(Abase + (tid >> 1) * K + kp + (tid & 1) * 16, 0, 1);
            __builtin_prefetch(Wbase + (tid >> 2) * K + kp + (tid & 3) * 8, 0, 1);
        }

        F8 a0 = load_afrag(&sA[cur][(wr * 32 + 0 * 16 + ml) * GEMM_LDSTR], kh);
        F8 a1 = load_afrag(&sA[cur][(wr * 32 + 1 * 16 + ml) * GEMM_LDSTR], kh);
        F8 b0 = load_bfrag(&sW[cur][(wc * 32 + 0 * 16 + ml) * GEMM_LDSTR], kh);
        F8 b1 = load_bfrag(&sW[cur][(wc * 32 + 1 * 16 + ml) * GEMM_LDSTR], kh);
        acc[0][0] = wmma_bf16(a0, b0, acc[0][0]);
        acc[0][1] = wmma_bf16(a0, b1, acc[0][1]);
        acc[1][0] = wmma_bf16(a1, b0, acc[1][0]);
        acc[1][1] = wmma_bf16(a1, b1, acc[1][1]);
        __syncthreads();
    }

    // Writeout: lane n = ml, VGPR r -> m = r + 8*kh
#pragma unroll
    for (int mi = 0; mi < 2; ++mi) {
#pragma unroll
        for (int ni = 0; ni < 2; ++ni) {
            unsigned int n = n0 + wc * 32 + ni * 16 + ml;
            float bv = HASBIAS ? bias[n] : 0.0f;
#pragma unroll
            for (int r = 0; r < 8; ++r) {
                unsigned int m = m0 + wr * 32 + mi * 16 + r + 8 * kh;
                float v = acc[mi][ni][r] * scale + bv;
                if (RELU) v = fmaxf(v, 0.0f);
                if (OUTMODE == 0)
                    ((float*)outp)[m * N + n] = v;
                else if (OUTMODE == 1)
                    ((unsigned short*)outp)[m * N + n] = f32_to_bf16(v);
                else
                    ((unsigned short*)outp)[n * M + m] = f32_to_bf16(v);
            }
        }
    }
}

// ---------------------------------------------------------------------------
// Flash-style causal attention.  One workgroup per (batch, 16-row Q tile),
// 8 waves each owning a 128-col output chunk.  Q,K row-major bf16 (Q holds
// the 1/sqrt(D) scale); V transposed bf16 [D][B*S] so P x V B-fragments are
// contiguous.  Output fp32 row-major.
// ---------------------------------------------------------------------------
__global__ void __launch_bounds__(256)
flash_attn_kernel(const unsigned short* __restrict__ Qb,
                  const unsigned short* __restrict__ Kb,
                  const unsigned short* __restrict__ Vt,
                  float* __restrict__ O,
                  unsigned int S, unsigned int D, unsigned int MS) {
    constexpr int QSTR = 1032;   // bf16 stride (mult of 8)
    constexpr int SSTR = 132;    // f32 stride
    constexpr int PSTR = 136;    // bf16 stride (mult of 8)

    __shared__ __align__(16) unsigned short sQ[16 * QSTR];
    __shared__ float                        sS[16 * SSTR];
    __shared__ __align__(16) unsigned short sP[16 * PSTR];
    __shared__ float sm[16], sl[16], sc[16];
    __shared__ float red[16 * 16];

    const unsigned int tid  = threadIdx.x;
    const int wave = tid >> 5;
    const int lane = tid & 31;
    const int kh   = lane >> 4;
    const int ml   = lane & 15;

    const unsigned int b  = blockIdx.y;
    const unsigned int q0 = blockIdx.x * 16u;
    const unsigned int bS = b * S;

    // Stage Q tile (16 x D bf16) into LDS with 128-bit copies
    {
        const v4u* src = reinterpret_cast<const v4u*>(Qb + (bS + q0) * D);
#pragma unroll
        for (int i = 0; i < 8; ++i) {
            unsigned int idx = i * 256u + tid;      // v4u index, 2048 total
            unsigned int r = idx >> 7, c = idx & 127;
            *reinterpret_cast<v4u*>(&sQ[r * QSTR + c * 8]) = src[r * 128 + c];
        }
    }
    if (tid < 16) { sm[tid] = -3.0e30f; sl[tid] = 0.0f; }

    v8f o[8];
#pragma unroll
    for (int nt = 0; nt < 8; ++nt) o[nt] = (v8f){};
    __syncthreads();

    const unsigned short* qrow = &sQ[ml * QSTR];
    const unsigned short* prow = &sP[ml * PSTR];

    const int nkb = (int)((q0 + 16 + 127) >> 7);
    for (int kb = 0; kb < nkb; ++kb) {
        const unsigned int kbase = (unsigned int)kb * 128u;

        // ---- Phase A: this wave's 16x16 score tile ----
        {
            const unsigned int kcol = kbase + wave * 16 + ml;
            const unsigned short* krow = Kb + (bS + kcol) * D;
            v8f s = (v8f){};
            for (unsigned int d0 = 0; d0 < D; d0 += 32) {
                F8 af = load_afrag(qrow + d0, kh);
                F8 bf = load_bfrag(krow + d0, kh);
                s = wmma_bf16(af, bf, s);
            }
#pragma unroll
            for (int r = 0; r < 8; ++r) {
                int mrow = r + 8 * kh;
                unsigned int kg = kbase + wave * 16 + ml;
                sS[mrow * SSTR + wave * 16 + ml] =
                    (kg <= q0 + (unsigned int)mrow) ? s[r] : -3.0e30f;
            }
        }
        __syncthreads();

        // ---- Phase B: online softmax over the 16x128 block ----
        const int row = tid >> 4;
        const int sub = tid & 15;
        {
            float lm = -3.0e30f;
#pragma unroll
            for (int j = 0; j < 8; ++j)
                lm = fmaxf(lm, sS[row * SSTR + sub * 8 + j]);
            red[row * 16 + sub] = lm;
        }
        __syncthreads();
        if (sub == 0) {
            float mb = red[row * 16];
#pragma unroll
            for (int j = 1; j < 16; ++j) mb = fmaxf(mb, red[row * 16 + j]);
            float mnew = fmaxf(sm[row], mb);
            sc[row] = __expf(sm[row] - mnew);
            sm[row] = mnew;
        }
        __syncthreads();
        {
            float mr = sm[row], ps = 0.0f;
#pragma unroll
            for (int j = 0; j < 8; ++j) {
                float p = __expf(sS[row * SSTR + sub * 8 + j] - mr);
                sP[row * PSTR + sub * 8 + j] = f32_to_bf16(p);
                ps += p;
            }
            red[row * 16 + sub] = ps;
        }
        __syncthreads();
        if (sub == 0) {
            float t = 0.0f;
#pragma unroll
            for (int j = 0; j < 16; ++j) t += red[row * 16 + j];
            sl[row] = sl[row] * sc[row] + t;
        }
        __syncthreads();

        // ---- Phase C: rescale O, accumulate P x V (V transposed) ----
        {
            // Prefetch next K block rows while P x V runs
            if (kb + 1 < nkb) {
                unsigned int knext = kbase + 128u + wave * 16 + ml;
                __builtin_prefetch(Kb + (bS + knext) * D, 0, 1);
            }

            float crv[8];
#pragma unroll
            for (int r = 0; r < 8; ++r) crv[r] = sc[r + 8 * kh];
#pragma unroll
            for (int nt = 0; nt < 8; ++nt)
#pragma unroll
                for (int r = 0; r < 8; ++r) o[nt][r] *= crv[r];

#pragma unroll
            for (int nt = 0; nt < 8; ++nt) {
                unsigned int ncol = wave * 128u + nt * 16u + ml;
                const unsigned short* vrow = Vt + ncol * MS + bS + kbase;
#pragma unroll
                for (int kc = 0; kc < 128; kc += 32) {
                    F8 af = load_afrag(prow + kc, kh);
                    F8 bf = load_bfrag(vrow + kc, kh);
                    o[nt] = wmma_bf16(af, bf, o[nt]);
                }
            }
        }
        __syncthreads();
    }

    // ---- Finalize: divide by l, write fp32 ----
    float invl[8];
#pragma unroll
    for (int r = 0; r < 8; ++r) invl[r] = 1.0f / sl[r + 8 * kh];
#pragma unroll
    for (int nt = 0; nt < 8; ++nt) {
        unsigned int ncol = wave * 128u + nt * 16u + ml;
#pragma unroll
        for (int r = 0; r < 8; ++r) {
            unsigned int mrow = r + 8 * kh;
            O[(bS + q0 + mrow) * D + ncol] = o[nt][r] * invl[r];
        }
    }
}

// ---------------------------------------------------------------------------
// Residual + LayerNorm: out[row] = LN(a[row] + b[row]) * gamma + beta
// ---------------------------------------------------------------------------
__global__ void __launch_bounds__(256)
residual_ln_kernel(const float* __restrict__ Xa, const float* __restrict__ Xb,
                   const float* __restrict__ gamma, const float* __restrict__ beta,
                   float* __restrict__ out, unsigned int D) {
    __shared__ float buf[1024];
    __shared__ float rs[256], rq[256];
    __shared__ float s_mu, s_inv;

    const unsigned int tid = threadIdx.x;
    const unsigned int base = blockIdx.x * D;

    float s = 0.0f, q = 0.0f;
#pragma unroll
    for (int i = 0; i < 4; ++i) {
        unsigned int c = i * 256u + tid;
        float v = Xa[base + c] + Xb[base + c];
        buf[c] = v;
        s += v;
        q += v * v;
    }
    rs[tid] = s; rq[tid] = q;
    __syncthreads();
    for (int off = 128; off > 0; off >>= 1) {
        if (tid < (unsigned)off) { rs[tid] += rs[tid + off]; rq[tid] += rq[tid + off]; }
        __syncthreads();
    }
    if (tid == 0) {
        float mu  = rs[0] / (float)D;
        float var = rq[0] / (float)D - mu * mu;
        s_mu  = mu;
        s_inv = rsqrtf(var + 1e-5f);
    }
    __syncthreads();
#pragma unroll
    for (int i = 0; i < 4; ++i) {
        unsigned int c = i * 256u + tid;
        out[base + c] = (buf[c] - s_mu) * s_inv * gamma[c] + beta[c];
    }
}

// ---------------------------------------------------------------------------
// Host launcher
// ---------------------------------------------------------------------------
extern "C" void kernel_launch(void* const* d_in, const int* in_sizes, int n_in,
                              void* d_out, int out_size, void* d_ws, size_t ws_size,
                              hipStream_t stream) {
    (void)in_sizes; (void)n_in; (void)out_size; (void)ws_size;

    constexpr unsigned int B = 4, S = 2048, D = 1024;
    constexpr unsigned int M = B * S;        // 8192 rows
    const float inv_sqrt_d = 0.03125f;       // 1/sqrt(1024)

    const float* x     = (const float*)d_in[0];
    const float* Wq    = (const float*)d_in[1];
    const float* Wk    = (const float*)d_in[2];
    const float* Wv    = (const float*)d_in[3];
    const float* W1    = (const float*)d_in[4];
    const float* b1    = (const float*)d_in[5];
    const float* W2    = (const float*)d_in[6];
    const float* b2    = (const float*)d_in[7];
    const float* gamma = (const float*)d_in[8];
    const float* beta  = (const float*)d_in[9];

    // Workspace layout (bytes):
    //   [0, 16M)   Qb bf16   -- later reused (with Kb) for ff2 f32
    //   [16M,32M)  Kb bf16
    //   [32M,48M)  Vt bf16 (transposed [D][M])
    //   [48M,80M)  attn f32  -- later reused for ff1
    //   [80M,112M) h f32
    char* ws = (char*)d_ws;
    const size_t MB = 1024ull * 1024ull;
    unsigned short* Qb  = (unsigned short*)(ws + 0 * MB);
    unsigned short* Kb  = (unsigned short*)(ws + 16 * MB);
    unsigned short* Vt  = (unsigned short*)(ws + 32 * MB);
    float*          att = (float*)(ws + 48 * MB);
    float*          h   = (float*)(ws + 80 * MB);
    float*          ff1 = att;
    float*          ff2 = (float*)(ws + 0 * MB);

    dim3 gGemm(D / 64, M / 128);
    dim3 bGemm(256);

    // Q/K: bf16 row-major (Q folds score scale).  V: bf16 transposed.
    gemm_bf16_kernel<1, false, false><<<gGemm, bGemm, 0, stream>>>(
        x, Wq, nullptr, Qb, M, D, D, inv_sqrt_d);
    gemm_bf16_kernel<1, false, false><<<gGemm, bGemm, 0, stream>>>(
        x, Wk, nullptr, Kb, M, D, D, 1.0f);
    gemm_bf16_kernel<2, false, false><<<gGemm, bGemm, 0, stream>>>(
        x, Wv, nullptr, Vt, M, D, D, 1.0f);

    flash_attn_kernel<<<dim3(S / 16, B), 256, 0, stream>>>(Qb, Kb, Vt, att, S, D, M);

    residual_ln_kernel<<<M, 256, 0, stream>>>(x, att, gamma, beta, h, D);

    gemm_bf16_kernel<0, true, true><<<gGemm, bGemm, 0, stream>>>(
        h, W1, b1, ff1, M, D, D, 1.0f);
    gemm_bf16_kernel<0, false, true><<<gGemm, bGemm, 0, stream>>>(
        ff1, W2, b2, ff2, M, D, D, 1.0f);

    residual_ln_kernel<<<M, 256, 0, stream>>>(h, ff2, gamma, beta, (float*)d_out, D);
}
